// Speller_75608604279510
// MI455X (gfx1250) — matrix operand order repaired and checked by reading, not compiled, not run on hardware
//
#include <hip/hip_runtime.h>
#include <hip/hip_bf16.h>

// ---------------------------------------------------------------------------
// Speller: teacher-forced 2-layer GRU decoder with dot attention.
// Persistent-kernel design: one launch runs all T=200 steps with grid-wide
// atomic barriers. Dense matmuls use v_wmma_f32_16x16x32_f16 (f32 accum);
// attention (batched matvec) uses f32 VALU + LDS softmax with vectorized
// float4 streaming from L2-resident encoder_outputs.
// ---------------------------------------------------------------------------

#define Bz   64
#define Tz   200
#define Dz   512
#define Sz   1024
#define Vz   30
#define NWG  32
#define HS   (Bz * Dz)      // 32768 state elements

typedef __attribute__((ext_vector_type(16))) _Float16 v16h;
typedef __attribute__((ext_vector_type(8)))  _Float16 v8h;
typedef __attribute__((ext_vector_type(4)))  _Float16 v4h;
typedef __attribute__((ext_vector_type(8)))  float    v8f;

// ---- WMMA fragment loaders (contiguous k = 16h + s order, same for A and B, wave32) ----------
// A (16x32, MxK): lanes 0-15 -> M=lane, elems 0..15 = K 0..15
//                 lanes 16-31 -> M=lane-16, elems 0..15 = K 16..31
__device__ __forceinline__ v16h load_a16(const _Float16* X, int ldx,
                                         int mbase, int kbase, int lane) {
  int m  = mbase + (lane & 15);
  int k0 = kbase + ((lane >> 4) << 4);
  const v8h* p = (const v8h*)(X + (size_t)m * ldx + k0);
  v8h lo = p[0];       // K k0..k0+7
  v8h hi = p[1];       // K k0+8..k0+15
  return __builtin_shufflevector(lo, hi, 0,1,2,3,4,5,6,7,8,9,10,11,12,13,14,15);
}

// B (32x16, KxN): lane -> N = lane&15; lanes 0-15 hold K 0..15, lanes 16-31 K 16..31.
// We need B[k,n] = W[nbase+n, kbase+k] with W row-major [N_total x K_total].
__device__ __forceinline__ v16h load_b16(const _Float16* W, int ldw,
                                         int nbase, int kbase, int lane) {
  int n  = nbase + (lane & 15);
  int k0 = kbase + ((lane >> 4) << 4);
  const v8h* p = (const v8h*)(W + (size_t)n * ldw + k0);
  v8h lo = p[0];
  v8h hi = p[1];
  return __builtin_shufflevector(lo, hi, 0,1,2,3,4,5,6,7,8,9,10,11,12,13,14,15);
}

#define WMMA(A, Bm, C) \
  __builtin_amdgcn_wmma_f32_16x16x32_f16(false, (A), false, (Bm), (short)0, (C), false, false)

__device__ __forceinline__ float sigmoidf_(float x) { return 1.0f / (1.0f + expf(-x)); }

// ---- Grid-wide barrier (sense via generation counter, device-scope) --------
__device__ __forceinline__ void grid_sync(unsigned* bar) {
  __syncthreads();
  if (threadIdx.x == 0) {
    __threadfence();
    unsigned gen = __hip_atomic_load(&bar[1], __ATOMIC_RELAXED, __HIP_MEMORY_SCOPE_AGENT);
    unsigned a   = __hip_atomic_fetch_add(&bar[0], 1u, __ATOMIC_ACQ_REL, __HIP_MEMORY_SCOPE_AGENT);
    if (a == (unsigned)(NWG - 1)) {
      __hip_atomic_store(&bar[0], 0u, __ATOMIC_RELAXED, __HIP_MEMORY_SCOPE_AGENT);
      __hip_atomic_fetch_add(&bar[1], 1u, __ATOMIC_RELEASE, __HIP_MEMORY_SCOPE_AGENT);
    } else {
      while (__hip_atomic_load(&bar[1], __ATOMIC_ACQUIRE, __HIP_MEMORY_SCOPE_AGENT) == gen)
        __builtin_amdgcn_s_sleep(1);
    }
    __threadfence();
  }
  __syncthreads();
}

// ---- Prep: f16 weight conversion, state init, barrier reset ----------------
__global__ void speller_prep(const float* __restrict__ Whh0, const float* __restrict__ Wih1,
                             const float* __restrict__ Whh1, const float* __restrict__ Wout,
                             const float* __restrict__ Wtok, const float* __restrict__ init_h,
                             _Float16* Whh0h, _Float16* Wih1h, _Float16* Whh1h,
                             _Float16* Wouth, _Float16* Wtokh,
                             float* h0f, float* h1f, _Float16* h0h, _Float16* h1h,
                             unsigned* bar) {
  int i = blockIdx.x * blockDim.x + threadIdx.x;
  if (i == 0) { bar[0] = 0u; bar[1] = 0u; }
  const int NW = 3 * Dz * Dz;                 // 1536*512
  if (i < NW) {
    Whh0h[i] = (_Float16)Whh0[i];
    Wih1h[i] = (_Float16)Wih1[i];
    Whh1h[i] = (_Float16)Whh1[i];
  }
  if (i < Dz * 2 * Dz) Wouth[i] = (_Float16)Wout[i];
  if (i < 32 * Dz) {
    int v = i >> 9, d = i & (Dz - 1);
    Wtokh[i] = (v < Vz) ? (_Float16)Wtok[v * Dz + d] : (_Float16)0.0f;
  }
  if (i < HS) {
    float a = init_h[i], b = init_h[HS + i];
    h0f[i] = a; h0h[i] = (_Float16)a;
    h1f[i] = b; h1h[i] = (_Float16)b;
  }
}

// ---- Main persistent kernel -------------------------------------------------
__global__ __launch_bounds__(256) void speller_main(
    const int*   __restrict__ targets,   // [B,T]
    const float* __restrict__ mem,       // [B,S,D] f32 (lives in L2)
    const float* __restrict__ Wih0,      // [3D,V] f32 (one-hot gather)
    const float* __restrict__ bih0, const float* __restrict__ bhh0,
    const float* __restrict__ bih1, const float* __restrict__ bhh1,
    const float* __restrict__ btok,
    const _Float16* __restrict__ Whh0h,  // [3D,D]
    const _Float16* __restrict__ Wih1h,  // [3D,D]
    const _Float16* __restrict__ Whh1h,  // [3D,D]
    const _Float16* __restrict__ Wouth,  // [D,2D]
    const _Float16* __restrict__ Wtokh,  // [32,D] (V padded to 32)
    float* h0f, float* h1f,              // [2][B,D] double-buffered masters
    _Float16* h0h, _Float16* h1h,        // [2][B,D] f16 copies
    _Float16* catb,                      // [B,2D]: cols 0..D-1 = c, D..2D-1 = q
    _Float16* attnh,                     // [B,D]
    float* out,                          // [B,T,V]
    unsigned* bar) {

  const int lane = threadIdx.x & 31;
  const int lw   = threadIdx.x >> 5;                 // wave in block (0..7)
  const int wid  = blockIdx.x * 8 + lw;              // global wave (0..255)

  __shared__ float sc[2][Sz];                        // softmax scores per local b
  __shared__ float cpart[2][4][Dz];                  // partial context sums

  for (int t = 0; t < Tz; ++t) {
    const int par = t & 1;
    const float*    h0f_r = h0f + par * HS;
    float*          h0f_w = h0f + (par ^ 1) * HS;
    const float*    h1f_r = h1f + par * HS;
    float*          h1f_w = h1f + (par ^ 1) * HS;
    const _Float16* h0h_r = h0h + par * HS;
    _Float16*       h0h_w = h0h + (par ^ 1) * HS;
    const _Float16* h1h_r = h1h + par * HS;
    _Float16*       h1h_w = h1h + (par ^ 1) * HS;

    // ---------------- Stage A: GRU layer 0 (+ logits of step t-1) ----------
    if (wid < 128) {
      const int mtile = wid >> 5, j = wid & 31;
      const int mbase = mtile * 16, dbase = j * 16;
      v8f aR = {}, aZ = {}, aN = {};
      for (int kk = 0; kk < 16; ++kk) {
        const int kb = kk * 32;
        v16h A    = load_a16(h0h_r, Dz, mbase, kb, lane);
        v16h Br   = load_b16(Whh0h, Dz, dbase,          kb, lane);
        v16h Bz16 = load_b16(Whh0h, Dz, Dz + dbase,     kb, lane);
        v16h Bn   = load_b16(Whh0h, Dz, 2 * Dz + dbase, kb, lane);
        aR = WMMA(A, Br, aR);
        aZ = WMMA(A, Bz16, aZ);
        aN = WMMA(A, Bn, aN);
      }
      const int mo = (lane >> 4) << 3;
      const int d  = dbase + (lane & 15);
      const int tprev = (t > 0) ? (t - 1) : 0;
#pragma unroll
      for (int r = 0; r < 8; ++r) {
        const int b = mbase + mo + r;
        const int tok = targets[b * Tz + tprev];
        float gir = Wih0[(size_t)d * Vz + tok]            + bih0[d];
        float giz = Wih0[(size_t)(Dz + d) * Vz + tok]     + bih0[Dz + d];
        float gin = Wih0[(size_t)(2 * Dz + d) * Vz + tok] + bih0[2 * Dz + d];
        float rr = sigmoidf_(gir + aR[r] + bhh0[d]);
        float zz = sigmoidf_(giz + aZ[r] + bhh0[Dz + d]);
        float nn = tanhf(gin + rr * (aN[r] + bhh0[2 * Dz + d]));
        float hold = h0f_r[b * Dz + d];
        float hnew = (1.0f - zz) * nn + zz * hold;
        h0f_w[b * Dz + d] = hnew;
        h0h_w[b * Dz + d] = (_Float16)hnew;
      }
    } else if (wid < 136 && t > 0) {
      // logits for step t-1: attn_h @ W_tok^T + b_tok   ([64 x 32pad], K=512)
      const int u2 = wid - 128;
      const int mbase = (u2 >> 1) * 16, nbase = (u2 & 1) * 16;
      v8f acc = {};
      for (int kk = 0; kk < 16; ++kk) {
        const int kb = kk * 32;
        v16h A  = load_a16(attnh, Dz, mbase, kb, lane);
        v16h Bm = load_b16(Wtokh, Dz, nbase, kb, lane);
        acc = WMMA(A, Bm, acc);
      }
      const int mo = (lane >> 4) << 3;
      const int v = nbase + (lane & 15);
#pragma unroll
      for (int r = 0; r < 8; ++r) {
        const int b = mbase + mo + r;
        if (v < Vz) out[((size_t)b * Tz + (t - 1)) * Vz + v] = acc[r] + btok[v];
      }
    }
    grid_sync(bar);

    // ---------------- Stage B: GRU layer 1 ---------------------------------
    if (wid < 128) {
      const int mtile = wid >> 5, j = wid & 31;
      const int mbase = mtile * 16, dbase = j * 16;
      v8f iR = {}, iZ = {}, iN = {}, gR = {}, gZ = {}, gN = {};
#pragma unroll 2
      for (int kk = 0; kk < 16; ++kk) {
        const int kb = kk * 32;
        v16h A0 = load_a16(h0h_w, Dz, mbase, kb, lane);   // new h0 (input)
        v16h A1 = load_a16(h1h_r, Dz, mbase, kb, lane);   // old h1 (hidden)
        v16h B0r = load_b16(Wih1h, Dz, dbase,          kb, lane);
        v16h B0z = load_b16(Wih1h, Dz, Dz + dbase,     kb, lane);
        v16h B0n = load_b16(Wih1h, Dz, 2 * Dz + dbase, kb, lane);
        v16h B1r = load_b16(Whh1h, Dz, dbase,          kb, lane);
        v16h B1z = load_b16(Whh1h, Dz, Dz + dbase,     kb, lane);
        v16h B1n = load_b16(Whh1h, Dz, 2 * Dz + dbase, kb, lane);
        iR = WMMA(A0, B0r, iR);  iZ = WMMA(A0, B0z, iZ);  iN = WMMA(A0, B0n, iN);
        gR = WMMA(A1, B1r, gR);  gZ = WMMA(A1, B1z, gZ);  gN = WMMA(A1, B1n, gN);
      }
      const int mo = (lane >> 4) << 3;
      const int d  = dbase + (lane & 15);
#pragma unroll
      for (int r = 0; r < 8; ++r) {
        const int b = mbase + mo + r;
        float rr = sigmoidf_(iR[r] + bih1[d]          + gR[r] + bhh1[d]);
        float zz = sigmoidf_(iZ[r] + bih1[Dz + d]     + gZ[r] + bhh1[Dz + d]);
        float nn = tanhf(iN[r] + bih1[2 * Dz + d] + rr * (gN[r] + bhh1[2 * Dz + d]));
        float hold = h1f_r[b * Dz + d];
        float hnew = (1.0f - zz) * nn + zz * hold;
        h1f_w[b * Dz + d] = hnew;
        h1h_w[b * Dz + d] = (_Float16)hnew;
        catb[b * 2 * Dz + Dz + d] = (_Float16)hnew;   // q half of [c|q]
      }
    }
    grid_sync(bar);

    // ---------------- Stage C: dot attention + softmax + context -----------
    {
      const int g   = lw >> 2;                  // local batch slot (0/1)
      const int sub = lw & 3;                   // quarter of S
      const int b   = blockIdx.x * 2 + g;
      const float* qrow = h1f_w + b * Dz;       // q = new h1 (f32)

      // scores: 8 accumulators per lane; one q float4 reused across 8 rows.
      {
        float acc[8];
#pragma unroll
        for (int i = 0; i < 8; ++i) acc[i] = 0.0f;
        const float* base = mem + ((size_t)b * Sz + sub * 256 + lane) * Dz;
        for (int dd = 0; dd < Dz; dd += 4) {
          const float4 qv = *(const float4*)(qrow + dd);
#pragma unroll
          for (int i = 0; i < 8; ++i) {
            const float4 mv = *(const float4*)(base + (size_t)(i * 32) * Dz + dd);
            acc[i] = fmaf(mv.x, qv.x, acc[i]);
            acc[i] = fmaf(mv.y, qv.y, acc[i]);
            acc[i] = fmaf(mv.z, qv.z, acc[i]);
            acc[i] = fmaf(mv.w, qv.w, acc[i]);
          }
        }
#pragma unroll
        for (int i = 0; i < 8; ++i) sc[g][sub * 256 + i * 32 + lane] = acc[i];
      }
      __syncthreads();

      // softmax over S=1024 (each wave redundantly reduces its b-row)
      float mx = -3.4e38f;
      for (int k = 0; k < 32; ++k) mx = fmaxf(mx, sc[g][lane + 32 * k]);
      for (int off = 16; off > 0; off >>= 1) mx = fmaxf(mx, __shfl_xor(mx, off, 32));
      float ssum = 0.0f;
      for (int k = 0; k < 32; ++k) ssum += expf(sc[g][lane + 32 * k] - mx);
      for (int off = 16; off > 0; off >>= 1) ssum += __shfl_xor(ssum, off, 32);
      const float inv = 1.0f / ssum;
      __syncthreads();
      for (int i = 0; i < 8; ++i) {
        const int s = sub * 256 + i * 32 + lane;
        sc[g][s] = expf(sc[g][s] - mx) * inv;
      }
      __syncthreads();

      // context: float4 per lane, d = lane*4 + 128*i (coalesced 512B/instr)
      {
        float4 cacc[4];
#pragma unroll
        for (int i = 0; i < 4; ++i) cacc[i] = make_float4(0.f, 0.f, 0.f, 0.f);
        for (int s = sub * 256; s < sub * 256 + 256; ++s) {
          const float a = sc[g][s];
          const float* mrow = mem + ((size_t)b * Sz + s) * Dz;
#pragma unroll
          for (int i = 0; i < 4; ++i) {
            const float4 mv = *(const float4*)(mrow + lane * 4 + 128 * i);
            cacc[i].x = fmaf(a, mv.x, cacc[i].x);
            cacc[i].y = fmaf(a, mv.y, cacc[i].y);
            cacc[i].z = fmaf(a, mv.z, cacc[i].z);
            cacc[i].w = fmaf(a, mv.w, cacc[i].w);
          }
        }
#pragma unroll
        for (int i = 0; i < 4; ++i)
          *(float4*)&cpart[g][sub][lane * 4 + 128 * i] = cacc[i];
      }
      __syncthreads();
      {
        // final reduce: wave `sub` owns d in [sub*128, sub*128+128)
        const int d = sub * 128 + lane * 4;
        float4 p0 = *(const float4*)&cpart[g][0][d];
        float4 p1 = *(const float4*)&cpart[g][1][d];
        float4 p2 = *(const float4*)&cpart[g][2][d];
        float4 p3 = *(const float4*)&cpart[g][3][d];
        v4h cv;
        cv[0] = (_Float16)(p0.x + p1.x + p2.x + p3.x);
        cv[1] = (_Float16)(p0.y + p1.y + p2.y + p3.y);
        cv[2] = (_Float16)(p0.z + p1.z + p2.z + p3.z);
        cv[3] = (_Float16)(p0.w + p1.w + p2.w + p3.w);
        *(v4h*)(catb + b * 2 * Dz + d) = cv;     // c half of [c|q]
      }
    }
    grid_sync(bar);

    // ---------------- Stage D: attn_h = tanh([c|q] @ W_out^T) --------------
    if (wid < 128) {
      const int mtile = wid >> 5, j = wid & 31;
      const int mbase = mtile * 16, dbase = j * 16;
      v8f acc = {};
      for (int kk = 0; kk < 32; ++kk) {                 // K = 2D = 1024
        const int kb = kk * 32;
        v16h A  = load_a16(catb, 2 * Dz, mbase, kb, lane);
        v16h Bm = load_b16(Wouth, 2 * Dz, dbase, kb, lane);
        acc = WMMA(A, Bm, acc);
      }
      const int mo = (lane >> 4) << 3;
      const int d  = dbase + (lane & 15);
#pragma unroll
      for (int r = 0; r < 8; ++r) {
        const int b = mbase + mo + r;
        attnh[b * Dz + d] = (_Float16)tanhf(acc[r]);
      }
    }
    grid_sync(bar);
  }

  // ---------------- Final logits (t = T-1) ----------------------------------
  if (wid < 8) {
    const int mbase = (wid >> 1) * 16, nbase = (wid & 1) * 16;
    v8f acc = {};
    for (int kk = 0; kk < 16; ++kk) {
      const int kb = kk * 32;
      v16h A  = load_a16(attnh, Dz, mbase, kb, lane);
      v16h Bm = load_b16(Wtokh, Dz, nbase, kb, lane);
      acc = WMMA(A, Bm, acc);
    }
    const int mo = (lane >> 4) << 3;
    const int v = nbase + (lane & 15);
#pragma unroll
    for (int r = 0; r < 8; ++r) {
      const int b = mbase + mo + r;
      if (v < Vz) out[((size_t)b * Tz + (Tz - 1)) * Vz + v] = acc[r] + btok[v];
    }
  }
}

// ---------------------------------------------------------------------------
extern "C" void kernel_launch(void* const* d_in, const int* in_sizes, int n_in,
                              void* d_out, int out_size, void* d_ws, size_t ws_size,
                              hipStream_t stream) {
  const float* init_h  = (const float*)d_in[0];
  const int*   targets = (const int*)d_in[1];
  const float* mem     = (const float*)d_in[2];
  const float* Wih0    = (const float*)d_in[3];
  const float* Whh0    = (const float*)d_in[4];
  const float* bih0    = (const float*)d_in[5];
  const float* bhh0    = (const float*)d_in[6];
  const float* Wih1    = (const float*)d_in[7];
  const float* Whh1    = (const float*)d_in[8];
  const float* bih1    = (const float*)d_in[9];
  const float* bhh1    = (const float*)d_in[10];
  const float* Wout    = (const float*)d_in[11];
  const float* Wtok    = (const float*)d_in[12];
  const float* btok    = (const float*)d_in[13];

  char* ws = (char*)d_ws;
  size_t o = 0;
  auto alloc = [&](size_t bytes) { size_t r = o; o = (o + bytes + 255) & ~(size_t)255; return r; };

  unsigned*  bar    = (unsigned*)(ws + alloc(256));
  _Float16*  Whh0h  = (_Float16*)(ws + alloc((size_t)3 * Dz * Dz * 2));
  _Float16*  Wih1h  = (_Float16*)(ws + alloc((size_t)3 * Dz * Dz * 2));
  _Float16*  Whh1h  = (_Float16*)(ws + alloc((size_t)3 * Dz * Dz * 2));
  _Float16*  Wouth  = (_Float16*)(ws + alloc((size_t)Dz * 2 * Dz * 2));
  _Float16*  Wtokh  = (_Float16*)(ws + alloc((size_t)32 * Dz * 2));
  float*     h0f    = (float*)   (ws + alloc((size_t)2 * HS * 4));
  float*     h1f    = (float*)   (ws + alloc((size_t)2 * HS * 4));
  _Float16*  h0h    = (_Float16*)(ws + alloc((size_t)2 * HS * 2));
  _Float16*  h1h    = (_Float16*)(ws + alloc((size_t)2 * HS * 2));
  _Float16*  catb   = (_Float16*)(ws + alloc((size_t)Bz * 2 * Dz * 2));
  _Float16*  attnhp = (_Float16*)(ws + alloc((size_t)Bz * Dz * 2));

  const int prepN = 3 * Dz * Dz;                 // 786432 (largest segment)
  speller_prep<<<(prepN + 255) / 256, 256, 0, stream>>>(
      Whh0, Wih1, Whh1, Wout, Wtok, init_h,
      Whh0h, Wih1h, Whh1h, Wouth, Wtokh,
      h0f, h1f, h0h, h1h, bar);

  speller_main<<<NWG, 256, 0, stream>>>(
      targets, mem, Wih0, bih0, bhh0, bih1, bhh1, btok,
      Whh0h, Wih1h, Whh1h, Wouth, Wtokh,
      h0f, h1f, h0h, h1h, catb, attnhp,
      (float*)d_out, bar);
}
